// DREAMReconstructor_12154757448304
// MI455X (gfx1250) — matrix operand order repaired and checked by reading, not compiled, not run on hardware
//
#include <hip/hip_runtime.h>
#include <math.h>

// ---------------------------------------------------------------------------
// DREAM recurrent reconstructor for MI455X (gfx1250), wave32 + WMMA bf16.
// B=64, T=256, D=256, H=512, R=16.
// 4 persistent workgroups x 256 threads (8 waves): 16 batch rows each.
// launch_bounds(256,1) -> 2 waves/SIMD -> up to 512 VGPRs/wave.
// Global traffic uses explicit address_space(1) pointers (global_load, not
// flat_load) so the DS pipe is reserved for A-fragment ds_load_b128s.
// A-fragments are register-blocked: loaded once per kt, reused per N-tile.
// All GEMMs: v_wmma_f32_16x16x32_bf16 (fp32 accumulate).
// ---------------------------------------------------------------------------

#define GLB __attribute__((address_space(1)))

typedef __attribute__((ext_vector_type(16))) __bf16        bf16x16;
typedef __attribute__((ext_vector_type(8)))  float         f32x8;
typedef __attribute__((ext_vector_type(4)))  float         f32x4;
typedef __attribute__((ext_vector_type(4)))  unsigned int  u32x4;

typedef GLB const unsigned short* gushort_cp;

struct U32x8 { u32x4 lo; u32x4 hi; };

__device__ __forceinline__ unsigned short f2bf(float f) {
  unsigned int u = __builtin_bit_cast(unsigned int, f);
  u += 0x7FFFu + ((u >> 16) & 1u);                  // round-to-nearest-even
  return (unsigned short)(u >> 16);
}
__device__ __forceinline__ float bf2f(unsigned short h) {
  unsigned int u = ((unsigned int)h) << 16;
  return __builtin_bit_cast(float, u);
}
__device__ __forceinline__ bf16x16 frag_cast(u32x4 lo, u32x4 hi) {
  U32x8 s; s.lo = lo; s.hi = hi;
  return __builtin_bit_cast(bf16x16, s);
}
// Opaque copy of a uniform pointer through an SGPR: defeats LICM/CSE so the
// dependent loads are re-issued every recurrence step instead of being
// hoisted out of the t-loop and spilled.  Emits no instruction.  Returns an
// address_space(1) pointer so loads lower to global_load (not flat_load).
__device__ __forceinline__ gushort_cp opq(const unsigned short* p) {
  unsigned long long u = (unsigned long long)p;
  asm volatile("" : "+s"(u));
  return (gushort_cp)u;
}
// A operand (16x32 bf16) from LDS: lane l -> row m=l&15; K range
// [2*ahi, 2*ahi+16) where ahi=(l>>4)*8.  rowk = row start + kt*32.
__device__ __forceinline__ bf16x16 load_afrag(const unsigned short* rowk, int ahi) {
  u32x4 lo = *(const u32x4*)(rowk + 2 * ahi);
  u32x4 hi = *(const u32x4*)(rowk + 2 * ahi + 8);
  return frag_cast(lo, hi);
}
// B operand: pre-packed per-lane order -> lane l reads 32 contiguous bytes.
__device__ __forceinline__ bf16x16 load_bfrag(gushort_cp p) {
  const GLB u32x4* q = (const GLB u32x4*)p;
  return frag_cast(q[0], q[1]);
}
__device__ __forceinline__ f32x8 wmma_bf16(bf16x16 a, bf16x16 b, f32x8 c) {
  return __builtin_amdgcn_wmma_f32_16x16x32_bf16(false, a, false, b,
                                                 (short)0, c, false, false);
}

// ---- weight packing: W[N,K] fp32 -> bf16 fragment order -------------------
// tile = nt*(K/32)+kt ; within tile: lane l (32), elem e (16):
//   value = W[nt*16 + (l&15)][kt*32 + (l>>4)*16 + e]
__global__ __launch_bounds__(256) void pack_bf16(const float* __restrict__ W,
                                                 unsigned short* __restrict__ o,
                                                 int N, int K) {
  int idx = blockIdx.x * 256 + threadIdx.x;
  if (idx >= N * K) return;
  int e    = idx & 15;
  int l    = (idx >> 4) & 31;
  int tile = idx >> 9;
  int KT   = K >> 5;
  int nt   = tile / KT;
  int kt   = tile - nt * KT;
  int n    = nt * 16 + (l & 15);
  int k    = kt * 32 + ((l >> 4) << 4) + e;
  o[idx]   = f2bf(W[(size_t)n * K + k]);
}

// ---------------------------------------------------------------------------
#define TT 256
#define DD 256
#define HH 512
#define RR 16
#define BLK 256   // 8 waves

__global__ __launch_bounds__(BLK, 1) void dream_main(
    const float* __restrict__ x, const float* __restrict__ b_dec,
    const unsigned short* __restrict__ Wpred_p,
    const unsigned short* __restrict__ Win_p,
    const unsigned short* __restrict__ Wrec_p,
    const unsigned short* __restrict__ Wdec_p,
    const unsigned short* __restrict__ Bproj_p,
    float* __restrict__ Ag, float* __restrict__ out) {
  __shared__ unsigned short s_hb[16 * HH];   // h (bf16), persistent
  __shared__ float          s_pool[16 * HH]; // overlays: x|xhat  OR  pre
  __shared__ unsigned short s_xb[16 * DD];   // x (bf16)
  __shared__ float          s_k[16 * 16];    // k = x @ B_proj^T
  __shared__ float          s_bdec[DD];
  __shared__ float s_err[16], s_e[16], s_s[16], s_plast[16], s_tauinv[16], s_sleep[16];

  float* const s_x    = s_pool;
  float* const s_xhat = s_pool + 16 * DD;
  float* const s_pre  = s_pool;

  const GLB float* xg   = (const GLB float*)x;
  GLB float*       outg = (GLB float*)out;
  GLB float*       AgG  = (GLB float*)Ag;

  const int tid   = threadIdx.x;
  const int wv    = tid >> 5;        // 0..7
  const int ln    = tid & 31;
  const int mbase = (ln >> 4) * 8;   // C/D frag: m = mbase + j
  const int ncol  = ln & 15;         // C/D frag: n = ntile*16 + ncol
  const int ahi   = (ln >> 4) * 8;   // A frag chunk offset
  const int arow  = ln & 15;         // A frag row
  const int b0    = blockIdx.x * 16;

  float hreg[4][8];                  // fp32 master h, tiles nt = wv + 8*tt
#pragma unroll
  for (int tt = 0; tt < 4; ++tt)
#pragma unroll
    for (int j = 0; j < 8; ++j) hreg[tt][j] = 0.f;

  for (int i = tid; i < 16 * HH; i += BLK) s_hb[i] = 0;
  if (tid < 16) { s_e[tid] = 0.f; s_s[tid] = 0.f; }
  s_bdec[tid] = b_dec[tid];
  __syncthreads();

  const int xm = tid >> 4;           // 0..15 row for x-load / err phases
  const int xd = (tid & 15) * 16;    // 16 consecutive d per thread

#pragma unroll 1
  for (int t = 0; t < TT; ++t) {
    // ---- P1: stage x_t (fp32 + bf16), reset err accumulators -------------
    {
      const GLB float* xp = xg + ((size_t)(b0 + xm) * TT + t) * DD + xd;
      float* dst = s_x + xm * DD + xd;
      unsigned short* db = s_xb + xm * DD + xd;
#pragma unroll
      for (int q = 0; q < 4; ++q) {
        f32x4 v = *(const GLB f32x4*)(xp + 4 * q);
        dst[4 * q + 0] = v.x; dst[4 * q + 1] = v.y;
        dst[4 * q + 2] = v.z; dst[4 * q + 3] = v.w;
        db[4 * q + 0] = f2bf(v.x); db[4 * q + 1] = f2bf(v.y);
        db[4 * q + 2] = f2bf(v.z); db[4 * q + 3] = f2bf(v.w);
      }
      if (tid < 16) s_err[tid] = 0.f;
    }
    __syncthreads();

    // ---- P2: x_hat = h @ Wpred^T (2 tiles/wave, A-frag reuse) ------------
    {
      gushort_cp Wp = opq(Wpred_p);
      f32x8 acc[2];
#pragma unroll
      for (int tt = 0; tt < 2; ++tt)
#pragma unroll
        for (int j = 0; j < 8; ++j) acc[tt][j] = 0.f;
#pragma unroll 4
      for (int kt = 0; kt < 16; ++kt) {
        bf16x16 a = load_afrag(s_hb + arow * HH + kt * 32, ahi);
#pragma unroll
        for (int tt = 0; tt < 2; ++tt) {
          const int nt = wv + 8 * tt;
          bf16x16 b = load_bfrag(Wp + (nt * 16 + kt) * 512 + ln * 16);
          acc[tt] = wmma_bf16(a, b, acc[tt]);
        }
      }
#pragma unroll
      for (int tt = 0; tt < 2; ++tt)
#pragma unroll
        for (int j = 0; j < 8; ++j)
          s_xhat[(mbase + j) * DD + (wv + 8 * tt) * 16 + ncol] = acc[tt][j];
      if (wv == 0) {                 // k = x @ Bproj^T (single tile)
        gushort_cp Bp = opq(Bproj_p);
        f32x8 ka;
#pragma unroll
        for (int j = 0; j < 8; ++j) ka[j] = 0.f;
#pragma unroll 4
        for (int kt = 0; kt < 8; ++kt) {
          bf16x16 a = load_afrag(s_xb + arow * DD + kt * 32, ahi);
          bf16x16 b = load_bfrag(Bp + kt * 512 + ln * 16);
          ka = wmma_bf16(a, b, ka);
        }
#pragma unroll
        for (int j = 0; j < 8; ++j) s_k[(mbase + j) * 16 + ncol] = ka[j];
      }
    }
    __syncthreads();

    // ---- P3: surprise / EMAs / plasticity / tau / sleep ------------------
    {
      float acc = 0.f;
      const float* xr = s_x + xm * DD + xd;
      const float* hr = s_xhat + xm * DD + xd;
#pragma unroll
      for (int i = 0; i < 16; ++i) { float d = xr[i] - hr[i]; acc += d * d; }
      atomicAdd(&s_err[xm], acc);
    }
    __syncthreads();
    if (tid < 16) {
      float err = s_err[tid] * (1.0f / (float)DD);
      float e   = s_e[tid];
      float sp  = 1.f / (1.f + expf(-(err - e) * 20.f));   // temp = 0.05
      s_e[tid] = 0.95f * e + 0.05f * err;
      s_s[tid] = 0.95f * s_s[tid] + 0.05f * sp;
      float pl = sp - 0.3f;
      s_plast[tid]  = pl > 0.f ? 0.5f * pl : 0.f;
      s_tauinv[tid] = (1.f + 5.f * sp) * 0.2f;             // 1/tau
      s_sleep[tid]  = sp < 0.15f ? 0.99f : 1.f;
    }
    __syncthreads();

    // ---- P5: fast = A.k ; write kappa*fast into s_pre (C operand) --------
    {
#pragma unroll 1
      for (int m = 0; m < 16; ++m) {
        const f32x4* kp = (const f32x4*)(s_k + m * 16);
        f32x4 k0 = kp[0], k1 = kp[1], k2 = kp[2], k3 = kp[3];
#pragma unroll
        for (int q = 0; q < 2; ++q) {
          const int hh = tid + q * BLK;
          const GLB f32x4* ap =
              (const GLB f32x4*)(AgG + ((size_t)(b0 + m) * HH + hh) * RR);
          f32x4 a0 = ap[0], a1 = ap[1], a2 = ap[2], a3 = ap[3];
          float dot = a0.x * k0.x + a0.y * k0.y + a0.z * k0.z + a0.w * k0.w
                    + a1.x * k1.x + a1.y * k1.y + a1.z * k1.z + a1.w * k1.w
                    + a2.x * k2.x + a2.y * k2.y + a2.z * k2.z + a2.w * k2.w
                    + a3.x * k3.x + a3.y * k3.y + a3.z * k3.z + a3.w * k3.w;
          s_pre[m * HH + hh] = 0.5f * dot;                 // kappa = 0.5
        }
      }
    }
    __syncthreads();

    // ---- P6: pre = C + x@Win^T + h@Wrec^T ; LTC h update -----------------
    {
      gushort_cp Wi = opq(Win_p);
      gushort_cp Wr = opq(Wrec_p);
      f32x8 c[4];
#pragma unroll
      for (int tt = 0; tt < 4; ++tt)
#pragma unroll
        for (int j = 0; j < 8; ++j)
          c[tt][j] = s_pre[(mbase + j) * HH + (wv + 8 * tt) * 16 + ncol];
#pragma unroll 2
      for (int kt = 0; kt < 8; ++kt) {
        bf16x16 a = load_afrag(s_xb + arow * DD + kt * 32, ahi);
#pragma unroll
        for (int tt = 0; tt < 4; ++tt) {
          const int nt = wv + 8 * tt;
          bf16x16 b = load_bfrag(Wi + (nt * 8 + kt) * 512 + ln * 16);
          c[tt] = wmma_bf16(a, b, c[tt]);
        }
      }
#pragma unroll 2
      for (int kt = 0; kt < 16; ++kt) {
        bf16x16 a = load_afrag(s_hb + arow * HH + kt * 32, ahi);
#pragma unroll
        for (int tt = 0; tt < 4; ++tt) {
          const int nt = wv + 8 * tt;
          bf16x16 b = load_bfrag(Wr + (nt * 16 + kt) * 512 + ln * 16);
          c[tt] = wmma_bf16(a, b, c[tt]);
        }
      }
      __syncthreads();                  // all h readers done before h write
#pragma unroll
      for (int tt = 0; tt < 4; ++tt)
#pragma unroll
        for (int j = 0; j < 8; ++j) {
          const int m = mbase + j;
          float ho = hreg[tt][j];
          float hn = ho + (tanhf(c[tt][j]) - ho) * s_tauinv[m];
          hreg[tt][j] = hn;
          s_hb[m * HH + (wv + 8 * tt) * 16 + ncol] = f2bf(hn);
        }
      __syncthreads();
    }

    // ---- P8: recon = h_new @ Wdec^T + b_dec -> out -----------------------
    {
      gushort_cp Wd = opq(Wdec_p);
      f32x8 acc[2];
#pragma unroll
      for (int tt = 0; tt < 2; ++tt) {
        float bv = s_bdec[(wv + 8 * tt) * 16 + ncol];
#pragma unroll
        for (int j = 0; j < 8; ++j) acc[tt][j] = bv;
      }
#pragma unroll 4
      for (int kt = 0; kt < 16; ++kt) {
        bf16x16 a = load_afrag(s_hb + arow * HH + kt * 32, ahi);
#pragma unroll
        for (int tt = 0; tt < 2; ++tt) {
          const int nt = wv + 8 * tt;
          bf16x16 b = load_bfrag(Wd + (nt * 16 + kt) * 512 + ln * 16);
          acc[tt] = wmma_bf16(a, b, acc[tt]);
        }
      }
#pragma unroll
      for (int tt = 0; tt < 2; ++tt)
#pragma unroll
        for (int j = 0; j < 8; ++j)
          outg[((size_t)(b0 + mbase + j) * TT + t) * DD + (wv + 8 * tt) * 16 + ncol] =
              acc[tt][j];
    }

    // ---- P9: Hebbian fast-weight update (thread owns A[:, hh, :]) --------
    {
#pragma unroll 1
      for (int m = 0; m < 16; ++m) {
        float pl0 = s_plast[m];
        float sl = s_sleep[m];
        const f32x4* kp = (const f32x4*)(s_k + m * 16);
        f32x4 k0 = kp[0], k1 = kp[1], k2 = kp[2], k3 = kp[3];
#pragma unroll
        for (int q = 0; q < 2; ++q) {
          const int hh = tid + q * BLK;
          float pl = pl0 * bf2f(s_hb[m * HH + hh]);
          GLB f32x4* ap = (GLB f32x4*)(AgG + ((size_t)(b0 + m) * HH + hh) * RR);
          f32x4 a0 = ap[0], a1 = ap[1], a2 = ap[2], a3 = ap[3];
          a0 = (a0 * 0.995f + k0 * pl) * sl;
          a1 = (a1 * 0.995f + k1 * pl) * sl;
          a2 = (a2 * 0.995f + k2 * pl) * sl;
          a3 = (a3 * 0.995f + k3 * pl) * sl;
          ap[0] = a0; ap[1] = a1; ap[2] = a2; ap[3] = a3;
        }
      }
    }
    __syncthreads();
  }
}

// ---------------------------------------------------------------------------
extern "C" void kernel_launch(void* const* d_in, const int* in_sizes, int n_in,
                              void* d_out, int out_size, void* d_ws, size_t ws_size,
                              hipStream_t stream) {
  (void)in_sizes; (void)n_in; (void)out_size; (void)ws_size;
  const float* x      = (const float*)d_in[0];
  const float* W_in   = (const float*)d_in[1];
  const float* W_rec  = (const float*)d_in[2];
  const float* W_pred = (const float*)d_in[3];
  const float* B_proj = (const float*)d_in[4];
  const float* W_dec  = (const float*)d_in[5];
  const float* b_dec  = (const float*)d_in[6];
  float* out = (float*)d_out;
  char*  ws  = (char*)d_ws;

  // ws layout (bytes):
  //   [0, 2MB)           A  fast weights fp32 [64][512][16]
  //   [2MB, +256KB)      Wpred packed bf16 (N=256,K=512)
  //   next 256KB         Win   packed bf16 (N=512,K=256)
  //   next 512KB         Wrec  packed bf16 (N=512,K=512)
  //   next 256KB         Wdec  packed bf16 (N=256,K=512)
  //   next 8KB           Bproj packed bf16 (N=16, K=256)
  float*          Ag      = (float*)ws;
  unsigned short* Wpred_p = (unsigned short*)(ws + 2097152);
  unsigned short* Win_p   = (unsigned short*)(ws + 2359296);
  unsigned short* Wrec_p  = (unsigned short*)(ws + 2621440);
  unsigned short* Wdec_p  = (unsigned short*)(ws + 3145728);
  unsigned short* Bproj_p = (unsigned short*)(ws + 3407872);

  hipMemsetAsync(Ag, 0, (size_t)64 * 512 * 16 * 4, stream);
  pack_bf16<<<(256 * 512 + 255) / 256, 256, 0, stream>>>(W_pred, Wpred_p, 256, 512);
  pack_bf16<<<(512 * 256 + 255) / 256, 256, 0, stream>>>(W_in,   Win_p,   512, 256);
  pack_bf16<<<(512 * 512 + 255) / 256, 256, 0, stream>>>(W_rec,  Wrec_p,  512, 512);
  pack_bf16<<<(256 * 512 + 255) / 256, 256, 0, stream>>>(W_dec,  Wdec_p,  256, 512);
  pack_bf16<<<(16 * 256 + 255) / 256,  256, 0, stream>>>(B_proj, Bproj_p, 16,  256);
  dream_main<<<4, BLK, 0, stream>>>(x, b_dec, Wpred_p, Win_p, Wrec_p, Wdec_p,
                                    Bproj_p, Ag, out);
}
